// Net_77292231459118
// MI455X (gfx1250) — matrix operand order repaired and checked by reading, not compiled, not run on hardware
//
// CDNA5 / gfx1250 implementation of the SE(3)-GNN reference (wave32, WMMA f16->f32).
//
// Key restructuring vs reference: the GConvSE3 message GEMM is hoisted past the
// (linear) mean-aggregation: out[n] = W3c · mean_e(yc[e] (x) h[src[e]]) + b3c · mean_e h
// turning 34 GFLOP of per-edge work into one [16384 x 1056] @ [1056 x 128] GEMM.
// All matrix contractions use v_wmma_f32_16x16x32_f16 with f32 accumulation.
// Weights for the small per-edge MLPs are pre-packed in LDS into the exact
// WMMA B-operand lane layout so each B fetch is one contiguous 32B load.
// The big GEMM streams its packed B matrix through LDS with the Tensor Data
// Mover (tensor_load_to_lds + s_wait_tensorcnt), double-buffered per k-step.

#include <hip/hip_runtime.h>
#include <math.h>

typedef _Float16 half_t;
typedef __attribute__((ext_vector_type(16))) _Float16 v16h;
typedef __attribute__((ext_vector_type(8)))  _Float16 v8h;
typedef __attribute__((ext_vector_type(8)))  float    v8f;
typedef __attribute__((ext_vector_type(4)))  unsigned int uint4v;
typedef __attribute__((ext_vector_type(8)))  int      int8v;
typedef __attribute__((ext_vector_type(4)))  int      int4v;

#define N_NODES 16384
#define DEG     8
#define N_EDGES (N_NODES * DEG)
#define NUM_AA  2048
#define F_IN    39
#define F_MID   32
#define F_OUT   128
#define F_ATT   8
#define EDGE_DIM 78
#define RAD_MID 32

// Big GEMM K: 1024 (i*32+m outer-product columns) + 32 (mean_h bias columns) = 33 k-blocks
#define KBIG    1056
#define NKB     33

struct Params {
  const float *f, *pos, *edge_attr;
  const int   *src, *dst, *aa;
  const float *vw1,*vb1,*vg1,*vbe1,*vw2,*vb2,*vg2,*vbe2,*vw3,*vb3;
  const float *kw1,*kb1,*kg1,*kbe1,*kw2,*kb2,*kg2,*kbe2,*kw3,*kb3;
  const float *wq,*wp,*gng,*gnb;
  const float *cw1,*cb1,*cg1,*cbe1,*cw2,*cb2,*cg2,*cbe2,*cw3,*cb3;
  const float *wself;
  float  *q, *scores, *vws, *outacc;   // ws (f32)
  half_t *ycws, *hws, *Bpk;            // ws (f16)
  float  *out;                         // d_out [2048,128]
};

// ---------------- WMMA helpers (layouts per CDNA5 ISA) -------------

__device__ __forceinline__ v8f wmma16(v16h a, v16h b, v8f c) {
  // D = A(16x32 f16) * B(32x16 f16) + C(16x16 f32)
  return __builtin_amdgcn_wmma_f32_16x16x32_f16(false, a, false, b, (short)0, c,
                                                false, false);
}

// A operand from row-major f16 [16][stride] tile: lane<16 -> K=kbase+{0..15},
// lane>=16 -> K=kbase+{16..31} (contiguous order, matches packed B).
__device__ __forceinline__ v16h load_a_lds(const half_t* base, int stride, int kbase, int lane) {
  int row  = lane & 15;
  int koff = (lane >> 4) * 16;
  const half_t* p = base + row * stride + kbase + koff;
  v8h lo = *(const v8h*)(p);
  v8h hi = *(const v8h*)(p + 8);
  return __builtin_shufflevector(lo, hi, 0,1,2,3,4,5,6,7,8,9,10,11,12,13,14,15);
}

// B operand from pre-packed block: per (kblock, ntile) store 32 lanes x 16 halfs
// contiguously; element t of lane L is B[k = 16*(L>>4)+t][n = L&15].
__device__ __forceinline__ v16h load_b_pk(const half_t* blk, int lane) {
  return *(const v16h*)(blk + lane * 16);
}

// C/D tile scatter: lane holds col = lane&15, rows r + 8*(lane>>4), r=0..7.
__device__ __forceinline__ void store_c_lds(float* buf, int stride, int nbase, v8f c, int lane) {
  int col  = nbase + (lane & 15);
  int rsel = (lane >> 4) * 8;
#pragma unroll
  for (int r = 0; r < 8; ++r) buf[(r + rsel) * stride + col] = c[r];
}

// LayerNorm(32ch, eps=1e-5) with pre-LN bias, then ReLU, emit f16.
__device__ __forceinline__ void ln_relu_row(const float* xin, const float* bias,
                                            const float* g, const float* be, half_t* outp) {
  float x[32]; float mu = 0.f;
#pragma unroll
  for (int c = 0; c < 32; ++c) { x[c] = xin[c] + bias[c]; mu += x[c]; }
  mu *= (1.f / 32.f);
  float var = 0.f;
#pragma unroll
  for (int c = 0; c < 32; ++c) { float d = x[c] - mu; var += d * d; }
  var *= (1.f / 32.f);
  float rs = rsqrtf(var + 1e-5f);
#pragma unroll
  for (int c = 0; c < 32; ++c) {
    float y = (x[c] - mu) * rs * g[c] + be[c];
    outp[c] = (half_t)(y > 0.f ? y : 0.f);
  }
}

// ---- Tensor Data Mover: 1-D contiguous copy global -> LDS (CDNA5 ISA) -----
// D# group0: count=1 | lds_addr | 57-bit global addr | type=2.
// D# group1: data_size=3 (8B), tensor_dim0 = tile_dim0 = n8 elements, dim1 = 1.
// Issue from one wave only (TDM ignores EXEC; tracked with TENSORcnt).

__device__ __forceinline__ void tdm_load_1d(unsigned lds_off, const void* gptr, unsigned n8) {
  unsigned long long ga = (unsigned long long)gptr;
  uint4v g0;
  g0.x = 1u;                                                  // count=1, user desc
  g0.y = lds_off;                                             // lds_addr [63:32]
  g0.z = (unsigned)(ga & 0xFFFFFFFFu);                        // global_addr[31:0]
  g0.w = (unsigned)((ga >> 32) & 0x01FFFFFFu) | (2u << 30);   // addr[56:32] | type=2
  int8v g1;
  g1[0] = (int)(3u << 16);                // data_size=8B; no multicast/pad/iterate
  g1[1] = (int)((n8 & 0xFFFFu) << 16);    // tensor_dim0[15:0]  at bits 63:48
  g1[2] = (int)(((n8 >> 16) & 0xFFFFu) | (1u << 16)); // dim0[31:16] | tensor_dim1=1
  g1[3] = (int)((n8 & 0xFFFFu) << 16);    // tile_dim0 at bits 127:112 (dim1 hi = 0)
  g1[4] = 1;                              // tile_dim1 = 1, tile_dim2 = 0
  g1[5] = (int)n8;                        // tensor_dim0_stride low
  g1[6] = 0;
  g1[7] = 0;
  int4v z4 = {0, 0, 0, 0};
#if defined(__clang_major__) && (__clang_major__ >= 23)
  int8v z8 = {0, 0, 0, 0, 0, 0, 0, 0};
  __builtin_amdgcn_tensor_load_to_lds(g0, g1, z4, z4, z8, 0);
#else
  __builtin_amdgcn_tensor_load_to_lds(g0, g1, z4, z4, 0);
#endif
}

// ---------------- K0: q = f @ wq^T ([N,8]) ------------------------------------

__global__ void k0_query(Params P) {
  int idx = blockIdx.x * blockDim.x + threadIdx.x;
  if (idx >= N_NODES * 8) return;
  int n = idx >> 3, o = idx & 7;
  float s = 0.f;
  for (int i = 0; i < F_IN; ++i) s += P.wq[o * F_IN + i] * P.f[(size_t)n * F_IN + i];
  P.q[idx] = s;
}

// ---------------- K_packB: pack [K=1056][N=128] big-GEMM weights --------------
// B[k=i*32+m][o] = cw3[(o*32+i)*32 + m];  B[1024+i][o] = cb3[o*32+i]

__global__ void k_packB(Params P) {
  int idx = blockIdx.x * blockDim.x + threadIdx.x;
  if (idx >= NKB * 8 * 512) return;
  int kb = idx / (8 * 512), r = idx % (8 * 512);
  int nt = r / 512, qq = r % 512, l = qq / 16, e = qq % 16;
  int k = kb * 32 + ((l >> 4) * 16) + e;
  int o = nt * 16 + (l & 15);
  float v;
  if (k < 1024) {
    int i = k >> 5, m = k & 31;
    v = P.cw3[((size_t)o * 32 + i) * 32 + m];
  } else {
    int i = k - 1024;
    v = P.cb3[(size_t)o * 32 + i];
  }
  P.Bpk[idx] = (half_t)v;
}

// ---------------- K1: edge pass (radial trunks + attention v/k + yc) ----------
// 256 threads = 8 waves, 16 edges per wave, grid = E/128 = 1024 blocks.
// Dynamic LDS (sizes all 32B multiples, carved in this exact order):

#define W1PK_B  (3*3*2*512*2)   // [trunk][kb(3)][nt(2)] packed, K pad 79->96
#define W2PK_B  (3*2*512*2)
#define W3PK_B  (20*512*2)      // per trunk: [nt(20)] K=32, f pad 312->320
#define LNP_B   (3*6*32*4)      // {b1,g1,be1,b2,g2,be2} per trunk
#define B3V_B   (320*4)
#define FEAT_B  (8*16*96*2)
#define HSRC_B  (8*16*40*4)
#define TMP_B   (8*16*32*4)
#define ACT_B   (8*16*32*2)
#define VK_B    (8*16*16*4)
#define K1_LDS  (W1PK_B + W2PK_B + 2*W3PK_B + LNP_B + 2*B3V_B + FEAT_B + HSRC_B + TMP_B + ACT_B + VK_B)

__global__ void __launch_bounds__(256) k1_edge(Params P) {
  extern __shared__ char smem[];
  char* sp = smem;
  half_t* w1pk  = (half_t*)sp; sp += W1PK_B;
  half_t* w2pk  = (half_t*)sp; sp += W2PK_B;
  half_t* w3pkv = (half_t*)sp; sp += W3PK_B;
  half_t* w3pkk = (half_t*)sp; sp += W3PK_B;
  float*  lnp   = (float*)sp;  sp += LNP_B;
  float*  b3v   = (float*)sp;  sp += B3V_B;
  float*  b3k   = (float*)sp;  sp += B3V_B;
  half_t* featA = (half_t*)sp; sp += FEAT_B;
  float*  hsrc  = (float*)sp;  sp += HSRC_B;
  float*  tmp   = (float*)sp;  sp += TMP_B;
  half_t* actA  = (half_t*)sp; sp += ACT_B;
  float*  vkacc = (float*)sp;  sp += VK_B;

  const int tid = threadIdx.x, lane = tid & 31, w = tid >> 5;

  const float* W1[3] = {P.vw1, P.kw1, P.cw1};
  const float* W2[3] = {P.vw2, P.kw2, P.cw2};
  const float* LNS[3][6] = {{P.vb1, P.vg1, P.vbe1, P.vb2, P.vg2, P.vbe2},
                            {P.kb1, P.kg1, P.kbe1, P.kb2, P.kg2, P.kbe2},
                            {P.cb1, P.cg1, P.cbe1, P.cb2, P.cg2, P.cbe2}};

  // ---- stage packed weights once per block ----
  for (int idx = tid; idx < 3*3*2*512; idx += 256) {
    int t = idx / 3072, r = idx % 3072;
    int kb = r / 1024, r2 = r % 1024, nt = r2 / 512, qq = r2 % 512;
    int l = qq / 16, e = qq % 16;
    int k = kb * 32 + (l >> 4) * 16 + e, n = nt * 16 + (l & 15);
    w1pk[idx] = (half_t)((k < EDGE_DIM + 1) ? W1[t][n * (EDGE_DIM + 1) + k] : 0.f);
  }
  for (int idx = tid; idx < 3*2*512; idx += 256) {
    int t = idx / 1024, r = idx % 1024, nt = r / 512, qq = r % 512;
    int l = qq / 16, e = qq % 16;
    int k = (l >> 4) * 16 + e, n = nt * 16 + (l & 15);
    w2pk[idx] = (half_t)W2[t][n * 32 + k];
  }
  for (int idx = tid; idx < 2*20*512; idx += 256) {
    int which = idx / (20*512), r = idx % (20*512);
    int nt = r / 512, qq = r % 512, l = qq / 16, e = qq % 16;
    int k = (l >> 4) * 16 + e, fcol = nt * 16 + (l & 15);
    const float* w3 = which ? P.kw3 : P.vw3;
    (which ? w3pkk : w3pkv)[r] = (half_t)((fcol < 312) ? w3[fcol * 32 + k] : 0.f);
  }
  for (int idx = tid; idx < 3*6*32; idx += 256) {
    int t = idx / 192, a = (idx % 192) / 32, c = idx % 32;
    lnp[idx] = LNS[t][a][c];
  }
  for (int idx = tid; idx < 640; idx += 256) {
    int which = idx / 320, fcol = idx % 320;
    float v = (fcol < 312) ? (which ? P.kb3[fcol] : P.vb3[fcol]) : 0.f;
    (which ? b3k : b3v)[fcol] = v;
  }

  // ---- stage per-wave 16-edge tile ----
  const int eb = blockIdx.x * 128 + w * 16;
  half_t* fA = featA + w * 16 * 96;
  float*  hs = hsrc  + w * 16 * 40;
  float*  tm = tmp   + w * 16 * 32;
  half_t* aA = actA  + w * 16 * 32;
  float*  vk = vkacc + w * 16 * 16;

  for (int idx = lane; idx < 16 * 96; idx += 32) {
    int r = idx / 96, c = idx % 96;
    float v = (c < EDGE_DIM) ? P.edge_attr[(size_t)(eb + r) * EDGE_DIM + c] : 0.f;
    fA[idx] = (half_t)v;
  }
  if (lane < 16) {  // feat[:,78] = edge length  (same-wave LDS ops are ordered)
    int e = eb + lane;
    int s = P.src[e], d = P.dst[e];
    float dx = P.pos[d*3+0] - P.pos[s*3+0];
    float dy = P.pos[d*3+1] - P.pos[s*3+1];
    float dz = P.pos[d*3+2] - P.pos[s*3+2];
    fA[lane * 96 + EDGE_DIM] = (half_t)sqrtf(dx*dx + dy*dy + dz*dz);
  }
  for (int idx = lane; idx < 16 * F_IN; idx += 32) {
    int r = idx / F_IN, c = idx % F_IN;
    hs[r * 40 + c] = P.f[(size_t)P.src[eb + r] * F_IN + c];
  }
  if (lane < 16) {
#pragma unroll
    for (int o = 0; o < 16; ++o) vk[lane * 16 + o] = 0.f;
  }
  __syncthreads();

  // ---- trunks: 0=v_rad, 1=k_rad, 2=c_rad (all consume the same feat tile) ----
  for (int t = 0; t < 3; ++t) {
    {  // L1: [16,96] @ [96,32]
      v8f c0 = {}, c1 = {};
      for (int kb = 0; kb < 3; ++kb) {
        v16h a = load_a_lds(fA, 96, kb * 32, lane);
        c0 = wmma16(a, load_b_pk(w1pk + (t * 6 + kb * 2 + 0) * 512, lane), c0);
        c1 = wmma16(a, load_b_pk(w1pk + (t * 6 + kb * 2 + 1) * 512, lane), c1);
      }
      store_c_lds(tm, 32, 0,  c0, lane);
      store_c_lds(tm, 32, 16, c1, lane);
    }
    __syncthreads();
    if (lane < 16)
      ln_relu_row(tm + lane*32, lnp + (t*6+0)*32, lnp + (t*6+1)*32, lnp + (t*6+2)*32, aA + lane*32);
    __syncthreads();
    {  // L2: [16,32] @ [32,32]
      v16h a = load_a_lds(aA, 32, 0, lane);
      v8f d0 = {}, d1 = {};
      d0 = wmma16(a, load_b_pk(w2pk + (t * 2 + 0) * 512, lane), d0);
      d1 = wmma16(a, load_b_pk(w2pk + (t * 2 + 1) * 512, lane), d1);
      store_c_lds(tm, 32, 0,  d0, lane);
      store_c_lds(tm, 32, 16, d1, lane);
    }
    __syncthreads();
    if (lane < 16)
      ln_relu_row(tm + lane*32, lnp + (t*6+3)*32, lnp + (t*6+4)*32, lnp + (t*6+5)*32, aA + lane*32);
    __syncthreads();

    if (t == 2) {
      // c_rad trunk output yc -> ws (consumed by k3_conv)
      for (int idx = lane; idx < 16 * 32; idx += 32)
        P.ycws[(size_t)(eb + idx / 32) * 32 + (idx & 31)] = aA[idx];
    } else {
      // v/k: T[16,320] = Y @ w3^T in 16-col chunks; contract columns f=o*39+i
      // against h_src on the fly:  v[e,o] += (T[e,f] + b3[f]) * h_src[e,i]
      const half_t* w3p = t ? w3pkk : w3pkv;
      const float*  b3p = t ? b3k   : b3v;
      v16h a = load_a_lds(aA, 32, 0, lane);
      const int col = lane & 15, rsel = (lane >> 4) * 8;
      for (int nt = 0; nt < 20; ++nt) {
        v8f cc = {};
        cc = wmma16(a, load_b_pk(w3p + nt * 512, lane), cc);
        int fidx = nt * 16 + col;
        if (fidx < 312) {
          int o = fidx / 39, i = fidx - o * 39;
          float bb = b3p[fidx];
#pragma unroll
          for (int r = 0; r < 8; ++r) {
            int row = r + rsel;
            atomicAdd(&vk[row * 16 + t * 8 + o], (cc[r] + bb) * hs[row * 40 + i]);
          }
        }
      }
    }
    __syncthreads();
  }

  // ---- attention logits + per-edge values -> ws ----
  if (lane < 16) {
    int e = eb + lane;
    int n = P.dst[e];
    float s = 0.f;
#pragma unroll
    for (int o = 0; o < 8; ++o) s += vk[lane * 16 + 8 + o] * P.q[(size_t)n * 8 + o];
    P.scores[e] = s * 0.35355339059327373f;  // 1/sqrt(F_ATT)
#pragma unroll
    for (int o = 0; o < 8; ++o) P.vws[(size_t)e * 8 + o] = vk[lane * 16 + o];
  }
}

// ---------------- K2: node pass (softmax, wp projection, GNorm, self) ---------
// dst = repeat(arange(N), 8) in this graph => edges of node n are [8n, 8n+8).

__global__ void __launch_bounds__(256) k2_node(Params P) {
  __shared__ __align__(32) half_t wppk[2 * 2 * 512];
  __shared__ __align__(32) half_t wspk[8 * 512];
  __shared__ float gng[32], gnb[32];
  __shared__ __align__(32) half_t catA[8][16 * 64];
  __shared__ float  tmp[8][16 * 32];
  __shared__ __align__(32) half_t hA[8][16 * 32];

  const int tid = threadIdx.x, lane = tid & 31, w = tid >> 5;

  for (int idx = tid; idx < 2 * 2 * 512; idx += 256) {
    int kb = idx / 1024, r = idx % 1024, nt = r / 512, qq = r % 512;
    int l = qq / 16, e = qq % 16;
    int k = kb * 32 + (l >> 4) * 16 + e, n = nt * 16 + (l & 15);
    wppk[idx] = (half_t)((k < F_ATT + F_IN) ? P.wp[n * (F_ATT + F_IN) + k] : 0.f);
  }
  for (int idx = tid; idx < 8 * 512; idx += 256) {
    int nt = idx / 512, qq = idx % 512, l = qq / 16, e = qq % 16;
    int k = (l >> 4) * 16 + e, n = nt * 16 + (l & 15);
    wspk[idx] = (half_t)P.wself[n * 32 + k];
  }
  if (tid < 32) { gng[tid] = P.gng[tid]; gnb[tid] = P.gnb[tid]; }

  const int nb = blockIdx.x * 128 + w * 16;
  if (lane < 16) {
    int n = nb + lane;
    float sc[8], mx = -3.4e38f;
#pragma unroll
    for (int j = 0; j < 8; ++j) { sc[j] = P.scores[n * 8 + j]; mx = fmaxf(mx, sc[j]); }
    float den = 0.f;
#pragma unroll
    for (int j = 0; j < 8; ++j) { sc[j] = expf(sc[j] - mx); den += sc[j]; }
    float inv = 1.f / den;
    float z[8] = {0, 0, 0, 0, 0, 0, 0, 0};
#pragma unroll
    for (int j = 0; j < 8; ++j) {
      float a = sc[j] * inv;
      const float* vp = P.vws + (size_t)(n * 8 + j) * 8;
#pragma unroll
      for (int o = 0; o < 8; ++o) z[o] += a * vp[o];
    }
    half_t* cr = catA[w] + lane * 64;
#pragma unroll
    for (int o = 0; o < 8; ++o) cr[o] = (half_t)z[o];
    for (int i = 0; i < F_IN; ++i) cr[8 + i] = (half_t)P.f[(size_t)n * F_IN + i];
    for (int c = 8 + F_IN; c < 64; ++c) cr[c] = (half_t)0.f;
  }
  __syncthreads();
  {  // h_pre = cat(z,f) @ wp^T : [16,64(pad47)] @ [64,32]
    v8f c0 = {}, c1 = {};
    for (int kb = 0; kb < 2; ++kb) {
      v16h a = load_a_lds(catA[w], 64, kb * 32, lane);
      c0 = wmma16(a, load_b_pk(wppk + (kb * 2 + 0) * 512, lane), c0);
      c1 = wmma16(a, load_b_pk(wppk + (kb * 2 + 1) * 512, lane), c1);
    }
    store_c_lds(tmp[w], 32, 0,  c0, lane);
    store_c_lds(tmp[w], 32, 16, c1, lane);
  }
  __syncthreads();
  if (lane < 16) {  // GNormSE3 (degree-0): relu(LN(|h|)) * sign(h)
    int n = nb + lane;
    float nrm[32], ph[32], mu = 0.f;
#pragma unroll
    for (int c = 0; c < 32; ++c) {
      float x = tmp[w][lane * 32 + c];
      float a = fmaxf(fabsf(x), 1e-12f);
      nrm[c] = a; ph[c] = x / a; mu += a;
    }
    mu *= (1.f / 32.f);
    float var = 0.f;
#pragma unroll
    for (int c = 0; c < 32; ++c) { float d = nrm[c] - mu; var += d * d; }
    var *= (1.f / 32.f);
    float rs = rsqrtf(var + 1e-5f);
#pragma unroll
    for (int c = 0; c < 32; ++c) {
      float y = (nrm[c] - mu) * rs * gng[c] + gnb[c];
      y = (y > 0.f ? y : 0.f) * ph[c];
      hA[w][lane * 32 + c] = (half_t)y;
      P.hws[(size_t)n * 32 + c] = (half_t)y;
    }
  }
  __syncthreads();
  {  // self-interaction: outacc = h @ w_self^T  ([16,32] @ [32,128])
    v16h a = load_a_lds(hA[w], 32, 0, lane);
    const int col = lane & 15, rsel = (lane >> 4) * 8;
#pragma unroll
    for (int nt = 0; nt < 8; ++nt) {
      v8f cc = {};
      cc = wmma16(a, load_b_pk(wspk + nt * 512, lane), cc);
#pragma unroll
      for (int r = 0; r < 8; ++r)
        P.outacc[(size_t)(nb + r + rsel) * 128 + nt * 16 + col] = cc[r];
    }
  }
}

// ---------------- K3: fused Cagg build + big GEMM (flagship) ------------------
// 128 threads = 4 waves x 16 nodes; LDS: 4 x [16][1056] f16 Cagg tiles (132 KB)
// + double-buffered 8 KB packed-B slice streamed by the TDM (16 KB) = 148 KB.

#define CAGG_B   (4 * 16 * KBIG * 2)
#define BBUF_B   (2 * 8 * 512 * 2)
#define K3_LDS   (CAGG_B + BBUF_B)

__global__ void __launch_bounds__(128) k3_conv(Params P) {
  extern __shared__ char smem[];
  half_t* Cagg = (half_t*)smem;
  half_t* Bbuf = (half_t*)(smem + CAGG_B);
  const unsigned bbuf_off = (unsigned)CAGG_B;  // LDS byte offset (dyn segment base 0)
  const int tid = threadIdx.x, lane = tid & 31, w = tid >> 5;
  half_t* CA = Cagg + (size_t)w * 16 * KBIG;
  const int nb = blockIdx.x * 64 + w * 16;

  // TDM: prefetch B k-slice 0 (8 KB = 1024 x 8B) into buffer 0 (wave 0 only).
  if (w == 0) tdm_load_1d(bbuf_off, P.Bpk, 1024);

  // Cagg[n, i*32+m] = (1/8) sum_e h[src[e],i]*yc[e,m];  Cagg[n,1024+i] = mean h_i
  for (int nl = 0; nl < 16; ++nl) {
    int n = nb + nl;
    float acc[32];
#pragma unroll
    for (int i = 0; i < 32; ++i) acc[i] = 0.f;
    float acc2 = 0.f;
    for (int j = 0; j < 8; ++j) {
      int e = n * 8 + j;
      int s = P.src[e];
      float hv = (float)P.hws[(size_t)s * 32 + lane];   // i = lane
      float yv = (float)P.ycws[(size_t)e * 32 + lane];  // m = lane
      acc2 += hv;
#pragma unroll
      for (int i = 0; i < 32; ++i)
        acc[i] += __shfl(hv, i, 32) * yv;               // broadcast h_i across wave
    }
    half_t* crow = CA + nl * KBIG;
#pragma unroll
    for (int i = 0; i < 32; ++i) crow[i * 32 + lane] = (half_t)(acc[i] * 0.125f);
    crow[1024 + lane] = (half_t)(acc2 * 0.125f);
  }

  // [16 x 1056] @ [1056 x 128]: 33 k-steps x 8 n-tiles = 264 WMMAs per wave.
  // B slices double-buffered through LDS via tensor_load_to_lds / TENSORcnt.
  v8f acc[8] = {};
  for (int kb = 0; kb < NKB; ++kb) {
    if (w == 0) __builtin_amdgcn_s_wait_tensorcnt(0);  // slice kb landed
    __syncthreads();                                   // visible to all waves
    if (w == 0 && kb + 1 < NKB)                        // prefetch slice kb+1
      tdm_load_1d(bbuf_off + (unsigned)(((kb + 1) & 1) * 8192),
                  P.Bpk + (size_t)(kb + 1) * 4096, 1024);
    const half_t* bb = Bbuf + (kb & 1) * 4096;
    v16h a = load_a_lds(CA, KBIG, kb * 32, lane);
#pragma unroll
    for (int nt = 0; nt < 8; ++nt)
      acc[nt] = wmma16(a, load_b_pk(bb + nt * 512, lane), acc[nt]);
    __syncthreads();   // all readers done before this buffer is overwritten again
  }
  const int col = lane & 15, rsel = (lane >> 4) * 8;
#pragma unroll
  for (int nt = 0; nt < 8; ++nt)
#pragma unroll
    for (int r = 0; r < 8; ++r) {
      float* p = &P.outacc[(size_t)(nb + r + rsel) * 128 + nt * 16 + col];
      *p += acc[nt][r];  // unique (row,col) per lane: race-free RMW
    }
}

// ---------------- K4: global_mean_pool over sorted aa -------------------------

__global__ void __launch_bounds__(128) k4_pool(Params P) {
  int g = blockIdx.x, t = threadIdx.x;
  int lo = 0, hi = N_NODES;
  while (lo < hi) { int mid = (lo + hi) >> 1; if (P.aa[mid] < g) lo = mid + 1; else hi = mid; }
  int start = lo;
  lo = start; hi = N_NODES;
  while (lo < hi) { int mid = (lo + hi) >> 1; if (P.aa[mid] < g + 1) lo = mid + 1; else hi = mid; }
  int end = lo;
  float s = 0.f;
  for (int r = start; r < end; ++r) s += P.outacc[(size_t)r * 128 + t];
  int c = end - start;
  P.out[(size_t)g * 128 + t] = s / (float)(c > 0 ? c : 1);
}

// ---------------- launcher ----------------------------------------------------

extern "C" void kernel_launch(void* const* d_in, const int* in_sizes, int n_in,
                              void* d_out, int out_size, void* d_ws, size_t ws_size,
                              hipStream_t stream) {
  (void)in_sizes; (void)n_in; (void)out_size; (void)ws_size;
  Params P;
  P.f = (const float*)d_in[0];  P.pos = (const float*)d_in[1];
  P.edge_attr = (const float*)d_in[2];
  P.src = (const int*)d_in[3];  P.dst = (const int*)d_in[4];  P.aa = (const int*)d_in[5];
  P.vw1=(const float*)d_in[6];  P.vb1=(const float*)d_in[7];  P.vg1=(const float*)d_in[8];  P.vbe1=(const float*)d_in[9];
  P.vw2=(const float*)d_in[10]; P.vb2=(const float*)d_in[11]; P.vg2=(const float*)d_in[12]; P.vbe2=(const float*)d_in[13];
  P.vw3=(const float*)d_in[14]; P.vb3=(const float*)d_in[15];
  P.kw1=(const float*)d_in[16]; P.kb1=(const float*)d_in[17]; P.kg1=(const float*)d_in[18]; P.kbe1=(const float*)d_in[19];
  P.kw2=(const float*)d_in[20]; P.kb2=(const float*)d_in[21]; P.kg2=(const float*)d_in[22]; P.kbe2=(const float*)d_in[23];
  P.kw3=(const float*)d_in[24]; P.kb3=(const float*)d_in[25];
  P.wq =(const float*)d_in[26]; P.wp =(const float*)d_in[27];
  P.gng=(const float*)d_in[28]; P.gnb=(const float*)d_in[29];
  P.cw1=(const float*)d_in[30]; P.cb1=(const float*)d_in[31]; P.cg1=(const float*)d_in[32]; P.cbe1=(const float*)d_in[33];
  P.cw2=(const float*)d_in[34]; P.cb2=(const float*)d_in[35]; P.cg2=(const float*)d_in[36]; P.cbe2=(const float*)d_in[37];
  P.cw3=(const float*)d_in[38]; P.cb3=(const float*)d_in[39];
  P.wself=(const float*)d_in[40];

  char* ws = (char*)d_ws;                                  // ~23 MB total
  P.q      = (float*)ws;  ws += (size_t)N_NODES * 8 * 4;
  P.scores = (float*)ws;  ws += (size_t)N_EDGES * 4;
  P.vws    = (float*)ws;  ws += (size_t)N_EDGES * 8 * 4;
  P.outacc = (float*)ws;  ws += (size_t)N_NODES * 128 * 4;
  P.ycws   = (half_t*)ws; ws += (size_t)N_EDGES * 32 * 2;
  P.hws    = (half_t*)ws; ws += (size_t)N_NODES * 32 * 2;
  P.Bpk    = (half_t*)ws; ws += (size_t)NKB * 8 * 512 * 2;
  P.out    = (float*)d_out;

  k0_query<<<(N_NODES * 8 + 255) / 256, 256, 0, stream>>>(P);
  k_packB <<<(NKB * 8 * 512 + 255) / 256, 256, 0, stream>>>(P);
  k1_edge <<<N_EDGES / 128, 256, (size_t)K1_LDS, stream>>>(P);   // 148 KB LDS: 2 WGs/WGP
  k2_node <<<N_NODES / 128, 256, 0, stream>>>(P);
  k3_conv <<<N_NODES / 64, 128, (size_t)K3_LDS, stream>>>(P);    // 148 KB LDS: 2 WGs/WGP
  k4_pool <<<NUM_AA, 128, 0, stream>>>(P);
}
